// Attention_7945689497706
// MI455X (gfx1250) — matrix operand order repaired and checked by reading, not compiled, not run on hardware
//
#include <hip/hip_runtime.h>

typedef __attribute__((ext_vector_type(16))) __bf16 v16bf;
typedef __attribute__((ext_vector_type(8)))  __bf16 v8bf;
typedef __attribute__((ext_vector_type(8)))  float  v8f;

#define HEADS    8
#define DIM      512
#define NPIX     1024      // 32*32
#define MKV      1056      // 4 mem slots + 1024 pixels, padded to 33*32
#define MKV_REAL 1028

__device__ __forceinline__ unsigned short f2bf(float f) {
  unsigned int u = __float_as_uint(f);
  u = u + 0x7FFFu + ((u >> 16) & 1u);          // round-to-nearest-even
  return (unsigned short)(u >> 16);
}

__device__ __forceinline__ v8bf ld8(const unsigned short* p) {
  return *(const v8bf*)p;
}
__device__ __forceinline__ v16bf cat16(v8bf a, v8bf b) {
  return __builtin_shufflevector(a, b, 0,1,2,3,4,5,6,7,8,9,10,11,12,13,14,15);
}
__device__ __forceinline__ v8f wmma_bf16(v16bf a, v16bf b, v8f c) {
  return __builtin_amdgcn_wmma_f32_16x16x32_bf16(false, a, false, b, (short)0, c,
                                                 false, false);
}
// A-operand (16x32): per lane two 16B chunks at k0+lh*16 and k0+lh*16+8
__device__ __forceinline__ v16bf loadA16(const unsigned short* arow, int k0, int lh) {
  return cat16(ld8(arow + k0 + lh * 16), ld8(arow + k0 + lh * 16 + 8));
}
// B-operand (32x16): per lane 16 consecutive K values
__device__ __forceinline__ v16bf loadB16(const unsigned short* bp) {
  return cat16(ld8(bp), ld8(bp + 8));
}

// ---------------------------------------------------------------- weights->bf16
__global__ __launch_bounds__(256) void convert_w_kernel(
    const float* __restrict__ wqkv, const float* __restrict__ wout,
    unsigned short* __restrict__ wqkv_bf, unsigned short* __restrict__ wout_bf) {
  int i = blockIdx.x * 256 + threadIdx.x;
  if (i < 3 * DIM * DIM) {
    wqkv_bf[i] = f2bf(wqkv[i]);
  } else {
    int j = i - 3 * DIM * DIM;
    if (j < DIM * DIM) wout_bf[j] = f2bf(wout[j]);
  }
}

// ---------------------------------------------------------------- RMSNorm -> bf16
// x: [B, 512, 1024] f32;   xn: [B, 1024, 512] bf16 (channel-contiguous)
__global__ __launch_bounds__(256) void rmsnorm_bf16_kernel(
    const float* __restrict__ x, const float* __restrict__ gamma,
    unsigned short* __restrict__ xn) {
  __shared__ float ssq[32];
  int tid = threadIdx.x;
  int b  = blockIdx.x >> 5;
  int n0 = (blockIdx.x & 31) << 5;
  if (tid < 32) ssq[tid] = 0.f;
  __syncthreads();
  int nl = tid & 31;       // pixel within tile (coalesced across lanes)
  int cg = tid >> 5;       // channel group 0..7
  const float* xb = x + (size_t)b * DIM * NPIX;
  float acc = 0.f;
  for (int k = 0; k < 64; ++k) {
    int c = cg + (k << 3);
    float v = xb[(size_t)c * NPIX + n0 + nl];
    acc += v * v;
  }
  atomicAdd(&ssq[nl], acc);
  __syncthreads();
  for (int idx = tid; idx < 32 * DIM; idx += 256) {
    int n = idx >> 9;
    int c = idx & (DIM - 1);
    float denom = fmaxf(sqrtf(ssq[n]), 1e-12f);
    float v = xb[(size_t)c * NPIX + n0 + n] / denom *
              (gamma[c] + 1.0f) * 22.627416997969522f;  // sqrt(512)
    xn[((size_t)b * NPIX + n0 + n) * DIM + c] = f2bf(v);
  }
}

// ---------------------------------------------------------------- mem_kv + padding
// kb: [B*H, 1056, 64] bf16 (row-major, d contig)   vb: [B*H, 64, 1056] bf16 (j contig)
__global__ __launch_bounds__(256) void fill_kv_kernel(
    const float* __restrict__ memkv, unsigned short* __restrict__ kb,
    unsigned short* __restrict__ vb) {
  int i = blockIdx.x * 256 + threadIdx.x;          // 524288 total
  if (i < 32768) {                                 // mem K slots
    int d = i & 63, m = (i >> 6) & 3, bh = i >> 8, h = bh & 7;
    kb[((size_t)bh * MKV + m) * 64 + d] = f2bf(memkv[(h * 4 + m) * 64 + d]);
  } else if (i < 65536) {                          // mem V slots (transposed)
    int j = i - 32768;
    int d = j & 63, m = (j >> 6) & 3, bh = j >> 8, h = bh & 7;
    vb[((size_t)bh * 64 + d) * MKV + m] = f2bf(memkv[2048 + (h * 4 + m) * 64 + d]);
  } else if (i < 65536 + 229376) {                 // K pad rows -> 0
    int j = i - 65536;
    int d = j & 63, r = (j >> 6) % 28, bh = j / (28 * 64);
    kb[((size_t)bh * MKV + MKV_REAL + r) * 64 + d] = 0;
  } else {                                         // V pad cols -> 0
    int j = i - 65536 - 229376;
    int d = j & 63, r = (j >> 6) % 28, bh = j / (28 * 64);
    vb[((size_t)bh * 64 + d) * MKV + MKV_REAL + r] = 0;
  }
}

// ---------------------------------------------------------------- QKV projection
// QKV[o,n] = W[o,c] * XN[c,n]; one wave -> 16(M) x 64(N) tile, K loop of 32,
// ping-pong double-buffered so loads of step k+1 overlap WMMAs of step k.
__global__ __launch_bounds__(256) void qkv_gemm_kernel(
    const unsigned short* __restrict__ wqkv_bf, const unsigned short* __restrict__ xn,
    unsigned short* __restrict__ qb, unsigned short* __restrict__ kb,
    unsigned short* __restrict__ vb) {
  int wg   = blockIdx.x * 8 + (threadIdx.x >> 5);
  int lane = threadIdx.x & 31, lm = lane & 15, lh = lane >> 4;
  int b   = wg / (96 * 16);
  int rem = wg % (96 * 16);
  int mt  = rem >> 4;             // 16-row output-channel tile (0..95)
  int n0  = (rem & 15) << 6;      // 64-wide pixel group
  const unsigned short* arow = wqkv_bf + (size_t)(mt * 16 + lm) * DIM;
  const unsigned short* brow = xn + ((size_t)b * NPIX + n0) * DIM;
  const unsigned short* bl   = brow + (size_t)lm * DIM + lh * 16;   // t stride 16*DIM

  v16bf A0, A1, B0[4], B1[4];
  A0 = loadA16(arow, 0, lh);
#pragma unroll
  for (int t = 0; t < 4; ++t) B0[t] = loadB16(bl + (size_t)t * 16 * DIM);
  v8f acc[4] = {};
#pragma unroll
  for (int kk = 0; kk < 16; kk += 2) {
    int k1 = (kk + 1) * 32;
    int k2 = (kk + 2 < 16) ? (kk + 2) * 32 : 0;   // clamped (redundant) tail load
    A1 = loadA16(arow, k1, lh);
#pragma unroll
    for (int t = 0; t < 4; ++t) B1[t] = loadB16(bl + (size_t)t * 16 * DIM + k1);
#pragma unroll
    for (int t = 0; t < 4; ++t) acc[t] = wmma_bf16(A0, B0[t], acc[t]);
    A0 = loadA16(arow, k2, lh);
#pragma unroll
    for (int t = 0; t < 4; ++t) B0[t] = loadB16(bl + (size_t)t * 16 * DIM + k2);
#pragma unroll
    for (int t = 0; t < 4; ++t) acc[t] = wmma_bf16(A1, B1[t], acc[t]);
  }

  int bh_base = b * HEADS;
#pragma unroll
  for (int t = 0; t < 4; ++t) {
    int n = n0 + t * 16 + lm;
#pragma unroll
    for (int r = 0; r < 8; ++r) {
      int o = mt * 16 + r + 8 * lh;      // C-layout: M = r + 8*(lane/16)
      float v = acc[t][r];
      if (o < DIM) {                     // Q, scaled by 1/sqrt(64), [n][d]
        int h = o >> 6, d = o & 63;
        qb[(((size_t)(bh_base + h)) * NPIX + n) * 64 + d] = f2bf(v * 0.125f);
      } else if (o < 2 * DIM) {          // K, [j][d], mem slots occupy j=0..3
        int oo = o - DIM, h = oo >> 6, d = oo & 63;
        kb[(((size_t)(bh_base + h)) * MKV + 4 + n) * 64 + d] = f2bf(v);
      } else {                           // V transposed, [d][j]
        int oo = o - 2 * DIM, h = oo >> 6, d = oo & 63;
        vb[(((size_t)(bh_base + h)) * 64 + d) * MKV + 4 + n] = f2bf(v);
      }
    }
  }
}

// ---------------------------------------------------------------- fused attention
// One wave per (b, h, 16 q-rows). Online softmax; P goes C-layout -> A-layout via LDS.
// V tiles loaded before the barrier; next K tiles prefetched across the PV WMMAs.
__global__ __launch_bounds__(256) void attn_kernel(
    const unsigned short* __restrict__ qb, const unsigned short* __restrict__ kb,
    const unsigned short* __restrict__ vb, unsigned short* __restrict__ ao) {
  __shared__ __attribute__((aligned(16))) unsigned short pl[8][16][32];
  int wid  = threadIdx.x >> 5;
  int lane = threadIdx.x & 31, lm = lane & 15, lh = lane >> 4;
  int task = blockIdx.x * 8 + wid;
  int bh = task >> 6;
  int n0 = (task & 63) << 4;
  const unsigned short* qrow = qb + ((size_t)bh * NPIX + n0 + lm) * 64;
  v16bf aq0 = cat16(ld8(qrow + lh * 16),      ld8(qrow + 8 + lh * 16));   // d 0..31
  v16bf aq1 = cat16(ld8(qrow + 32 + lh * 16), ld8(qrow + 40 + lh * 16));  // d 32..63
  v8f m_run, l_run;
  v8f accO[4] = {};
  for (int r = 0; r < 8; ++r) { m_run[r] = -1e30f; l_run[r] = 0.f; }
  const unsigned short* kbase = kb + (size_t)bh * MKV * 64 + (size_t)lm * 64 + lh * 16;
  const unsigned short* vbase = vb + (size_t)bh * 64 * MKV + (size_t)lm * MKV + lh * 16;

  v16bf kc[4];                           // [st*2+c]: sim B tiles, c = d-half
#pragma unroll
  for (int st = 0; st < 2; ++st) {
    const unsigned short* kp = kbase + (size_t)(st * 16) * 64;
    kc[st * 2 + 0] = loadB16(kp);
    kc[st * 2 + 1] = loadB16(kp + 32);
  }

  for (int j0 = 0; j0 < MKV; j0 += 32) {               // 33 iterations, uniform
    v8f s[2] = {};
#pragma unroll
    for (int st = 0; st < 2; ++st) {                   // sim tiles (n x 16 cols)
      s[st] = wmma_bf16(aq0, kc[st * 2 + 0], s[st]);
      s[st] = wmma_bf16(aq1, kc[st * 2 + 1], s[st]);
    }
#pragma unroll
    for (int st = 0; st < 2; ++st) {                   // mask padded KV columns
      int jcol = j0 + st * 16 + lm;
      if (jcol >= MKV_REAL)
        for (int r = 0; r < 8; ++r) s[st][r] = -1e30f;
    }
    v8f rmax;
    for (int r = 0; r < 8; ++r) rmax[r] = fmaxf(s[0][r], s[1][r]);
    for (int off = 1; off < 16; off <<= 1)             // reduce within 16-lane half
      for (int r = 0; r < 8; ++r) rmax[r] = fmaxf(rmax[r], __shfl_xor(rmax[r], off, 32));
    v8f m_new, alpha, p0, p1, rs;
    for (int r = 0; r < 8; ++r) {
      m_new[r] = fmaxf(m_run[r], rmax[r]);
      alpha[r] = __expf(m_run[r] - m_new[r]);
      p0[r] = __expf(s[0][r] - m_new[r]);
      p1[r] = __expf(s[1][r] - m_new[r]);
      rs[r] = p0[r] + p1[r];
    }
    for (int off = 1; off < 16; off <<= 1)
      for (int r = 0; r < 8; ++r) rs[r] += __shfl_xor(rs[r], off, 32);
    for (int r = 0; r < 8; ++r) {
      l_run[r] = l_run[r] * alpha[r] + rs[r];
      m_run[r] = m_new[r];
    }
#pragma unroll
    for (int t4 = 0; t4 < 4; ++t4) accO[t4] *= alpha;

    v16bf Bv[4];                                       // load V before the barrier
#pragma unroll
    for (int t4 = 0; t4 < 4; ++t4)
      Bv[t4] = loadB16(vbase + (size_t)(t4 * 16) * MKV + j0);

    // P: C-layout -> LDS (row, col) -> A-layout
#pragma unroll
    for (int r = 0; r < 8; ++r) {
      int row = r + 8 * lh;
      pl[wid][row][lm]      = f2bf(p0[r]);
      pl[wid][row][16 + lm] = f2bf(p1[r]);
    }
    __syncthreads();
    v16bf ap = cat16(ld8(&pl[wid][lm][lh * 16]), ld8(&pl[wid][lm][lh * 16 + 8]));

    int jn = (j0 + 32 < MKV) ? j0 + 32 : 0;            // prefetch next K tiles
#pragma unroll
    for (int st = 0; st < 2; ++st) {
      const unsigned short* kp = kbase + (size_t)(jn + st * 16) * 64;
      kc[st * 2 + 0] = loadB16(kp);
      kc[st * 2 + 1] = loadB16(kp + 32);
    }
#pragma unroll
    for (int t4 = 0; t4 < 4; ++t4)                     // PV: N = 4 d-tiles
      accO[t4] = wmma_bf16(ap, Bv[t4], accO[t4]);
    __syncthreads();
  }

  int b = bh >> 3, h = bh & 7;
  v8f inv;
  for (int r = 0; r < 8; ++r) inv[r] = 1.0f / l_run[r];
#pragma unroll
  for (int t4 = 0; t4 < 4; ++t4)
#pragma unroll
    for (int r = 0; r < 8; ++r) {
      int n = n0 + r + 8 * lh;
      int c = h * 64 + t4 * 16 + lm;
      ao[((size_t)b * NPIX + n) * DIM + c] = f2bf(accO[t4][r] * inv[r]);
    }
}

// ---------------------------------------------------------------- output projection
__global__ __launch_bounds__(256) void out_gemm_kernel(
    const unsigned short* __restrict__ wout_bf, const unsigned short* __restrict__ ao,
    float* __restrict__ out) {
  int wg   = blockIdx.x * 8 + (threadIdx.x >> 5);
  int lane = threadIdx.x & 31, lm = lane & 15, lh = lane >> 4;
  int b   = wg / (32 * 16);
  int rem = wg % (32 * 16);
  int mt  = rem >> 4;
  int n0  = (rem & 15) << 6;
  const unsigned short* arow = wout_bf + (size_t)(mt * 16 + lm) * DIM;
  const unsigned short* brow = ao + ((size_t)b * NPIX + n0) * DIM;
  const unsigned short* bl   = brow + (size_t)lm * DIM + lh * 16;

  v16bf A0, A1, B0[4], B1[4];
  A0 = loadA16(arow, 0, lh);
#pragma unroll
  for (int t = 0; t < 4; ++t) B0[t] = loadB16(bl + (size_t)t * 16 * DIM);
  v8f acc[4] = {};
#pragma unroll
  for (int kk = 0; kk < 16; kk += 2) {
    int k1 = (kk + 1) * 32;
    int k2 = (kk + 2 < 16) ? (kk + 2) * 32 : 0;
    A1 = loadA16(arow, k1, lh);
#pragma unroll
    for (int t = 0; t < 4; ++t) B1[t] = loadB16(bl + (size_t)t * 16 * DIM + k1);
#pragma unroll
    for (int t = 0; t < 4; ++t) acc[t] = wmma_bf16(A0, B0[t], acc[t]);
    A0 = loadA16(arow, k2, lh);
#pragma unroll
    for (int t = 0; t < 4; ++t) B0[t] = loadB16(bl + (size_t)t * 16 * DIM + k2);
#pragma unroll
    for (int t = 0; t < 4; ++t) acc[t] = wmma_bf16(A1, B1[t], acc[t]);
  }
#pragma unroll
  for (int t = 0; t < 4; ++t) {
    int n = n0 + t * 16 + lm;
#pragma unroll
    for (int r = 0; r < 8; ++r) {
      int o = mt * 16 + r + 8 * lh;
      out[((size_t)b * DIM + o) * NPIX + n] = acc[t][r];
    }
  }
}

// ---------------------------------------------------------------- launch
extern "C" void kernel_launch(void* const* d_in, const int* in_sizes, int n_in,
                              void* d_out, int out_size, void* d_ws, size_t ws_size,
                              hipStream_t stream) {
  const float* x     = (const float*)d_in[0];
  const float* gamma = (const float*)d_in[1];
  const float* memkv = (const float*)d_in[2];
  const float* wqkv  = (const float*)d_in[3];
  const float* wout  = (const float*)d_in[4];
  char* ws = (char*)d_ws;
  unsigned short* xn      = (unsigned short*)(ws);              // 16 MB
  unsigned short* wqkv_bf = (unsigned short*)(ws + 16777216);   // 1.5 MB
  unsigned short* wout_bf = (unsigned short*)(ws + 18350080);   // 0.5 MB
  unsigned short* qb      = (unsigned short*)(ws + 18874368);   // 16 MB
  unsigned short* kb      = (unsigned short*)(ws + 35651584);   // 16.5 MB
  unsigned short* vb      = (unsigned short*)(ws + 52953088);   // 16.5 MB
  unsigned short* ao      = (unsigned short*)(ws + 70254592);   // 16 MB
  float* out = (float*)d_out;

  convert_w_kernel<<<4096, 256, 0, stream>>>(wqkv, wout, wqkv_bf, wout_bf);
  rmsnorm_bf16_kernel<<<512, 256, 0, stream>>>(x, gamma, xn);
  fill_kv_kernel<<<2048, 256, 0, stream>>>(memkv, kb, vb);
  qkv_gemm_kernel<<<3072, 256, 0, stream>>>(wqkv_bf, xn, qb, kb, vb);
  attn_kernel<<<1024, 256, 0, stream>>>(qb, kb, vb, ao);
  out_gemm_kernel<<<1024, 256, 0, stream>>>(wout_bf, ao, out);
}
